// MultiheadAttention_42614665511356
// MI455X (gfx1250) — matrix operand order repaired and checked by reading, not compiled, not run on hardware
//
#include <hip/hip_runtime.h>

// ---------------------------------------------------------------------------
// MultiheadAttention for MI455X (gfx1250, wave32, WMMA bf16 16x16x32 + TDM)
// B=2, S=2048, E=2048, H=16, HD=128
// ---------------------------------------------------------------------------

typedef __attribute__((ext_vector_type(16))) __bf16 v16bf;
typedef __attribute__((ext_vector_type(8)))  __bf16 bf16x8;
typedef __attribute__((ext_vector_type(4)))  __bf16 bf16x4;
typedef __attribute__((ext_vector_type(8)))  float  v8f;
typedef __attribute__((ext_vector_type(4)))  unsigned int u32x4;
typedef __attribute__((ext_vector_type(4)))  int i32x4;
typedef __attribute__((ext_vector_type(8)))  int i32x8;

#define BATCH 2
#define S_LEN 2048
#define EMB   2048
#define NH    16
#define HDIM  128

// ---------------- WMMA fragment helpers (layouts per ISA 7.12.2) -----------
static __device__ inline v16bf load_a16(const __bf16* base, int ldk, int lane) {
  const int m = lane & 15, hf = (lane >> 4) & 1;
  const __bf16* p = base + m * ldk + hf * 8;
  bf16x8 lo = *(const bf16x8*)(p);
  bf16x8 hi = *(const bf16x8*)(p + 16);
  return __builtin_shufflevector(lo, hi, 0,1,2,3,4,5,6,7,8,9,10,11,12,13,14,15);
}

static __device__ inline v16bf load_a16_f32(const float* base, int ldk, int lane) {
  const int m = lane & 15, hf = (lane >> 4) & 1;
  const float* p0 = base + m * ldk + hf * 8;
  const float* p1 = p0 + 16;
  v16bf a;
#pragma unroll
  for (int j = 0; j < 8; ++j) { a[j] = (__bf16)p0[j]; a[j + 8] = (__bf16)p1[j]; }
  return a;
}

static __device__ inline v16bf load_b16(const __bf16* base, int ldk, int lane) {
  const int n = lane & 15, hf = (lane >> 4) & 1;
  const __bf16* p = base + n * ldk + hf * 8;
  bf16x8 lo = *(const bf16x8*)(p);
  bf16x8 hi = *(const bf16x8*)(p + 16);
  return __builtin_shufflevector(lo, hi, 0,1,2,3,4,5,6,7,8,9,10,11,12,13,14,15);
}

static __device__ inline v8f wmma_bf16(v16bf a, v16bf b, v8f c) {
  return __builtin_amdgcn_wmma_f32_16x16x32_bf16(
      /*neg_a=*/false, a, /*neg_b=*/false, b,
      /*c_mod=*/(short)0, c, /*reuse_a=*/false, /*reuse_b=*/false);
}

// ---------------------------------------------------------------------------
// Tensor Data Mover: 2D tile of bf16 (2-byte elems), global -> LDS, with LDS
// row padding. Descriptor layout per CDNA5 ISA.
//   group0: [1:0]=count=1, [63:32]=lds_addr, [120:64]=global_addr, [127:126]=2
//   group1: data_size=1(2B), pad_enable, pad_interval/pad_amount codes,
//           tensor_dim0/1, tile_dim0/1, tensor_dim0_stride (48b, elems)
// Issue from ONE wave (TDM ignores EXEC, one issue per wave); that wave must
// s_wait_tensorcnt 0 before the block barrier.
// ---------------------------------------------------------------------------
static __device__ inline void tdm_load_2d_bf16(unsigned lds_addr,
                                               const void* gptr,
                                               unsigned tile_w_elems,
                                               unsigned tile_h_rows,
                                               unsigned row_stride_elems,
                                               unsigned pad_interval_code,
                                               unsigned pad_amount_code) {
  unsigned long long ga = (unsigned long long)gptr;

  u32x4 g0;
  g0[0] = 1u;                                        // count=1 (valid), user D#
  g0[1] = lds_addr;                                  // LDS byte address
  g0[2] = (unsigned)(ga & 0xffffffffu);              // global_addr[31:0]
  g0[3] = (unsigned)((ga >> 32) & 0x01ffffffu)       // global_addr[56:32]
        | (2u << 30);                                // type = 2 ("image")

  const unsigned td0 = tile_w_elems;                 // tensor_dim0
  const unsigned td1 = tile_h_rows;                  // tensor_dim1 (tile fits)
  const unsigned long long st0 = row_stride_elems;   // tensor_dim0_stride

  i32x8 g1;
  g1[0] = (int)((1u << 16)                           // data_size = 1 (2 bytes)
              | (1u << 20)                           // pad_enable
              | (pad_interval_code << 22)
              | (pad_amount_code << 25));
  g1[1] = (int)((td0 & 0xffffu) << 16);              // ab_addr=0 | td0[15:0]
  g1[2] = (int)((td0 >> 16) | ((td1 & 0xffffu) << 16));
  g1[3] = (int)((td1 >> 16) | (tile_w_elems << 16)); // tile_dim0
  g1[4] = (int)(tile_h_rows & 0xffffu);              // tile_dim1 | tile_dim2=0
  g1[5] = (int)(st0 & 0xffffffffu);
  g1[6] = (int)((st0 >> 32) & 0xffffu);              // stride0 hi | stride1 lo=0
  g1[7] = 0;

  i32x4 z4 = {0, 0, 0, 0};
#if defined(__clang_major__) && (__clang_major__ >= 23)
  i32x8 z8 = {0, 0, 0, 0, 0, 0, 0, 0};
  __builtin_amdgcn_tensor_load_to_lds(g0, g1, z4, z4, z8, 0);
#else
  __builtin_amdgcn_tensor_load_to_lds(g0, g1, z4, z4, 0);
#endif
}

// ---------------------------------------------------------------------------
// Kernel 1/4: tiled GEMM C[MxN] = A[MxK] * B[KxN]
// ---------------------------------------------------------------------------
#define BM 128
#define BN 128
#define BK 32
#define LDT 40  // BK + 8 pad (80B pitch, 16B aligned)

template <bool A_F32, bool OUT_BF16>
__global__ __launch_bounds__(256) void gemm_tile(const void* __restrict__ Ap,
                                                 const float* __restrict__ Bp,
                                                 void* __restrict__ Cp,
                                                 int M, int N, int K) {
  __shared__ __align__(16) __bf16 As[BM][LDT];
  __shared__ __align__(16) __bf16 Bs[BN][LDT];

  const int tid  = threadIdx.x;
  const int lane = tid & 31;
  const int w    = tid >> 5;
  const int wm   = (w >> 2) * 64;
  const int wn   = (w & 3) * 32;
  const int m0   = blockIdx.y * BM;
  const int n0   = blockIdx.x * BN;
  const int hf   = lane >> 4;
  const int nl   = lane & 15;

  v8f acc[4][2];
#pragma unroll
  for (int mt = 0; mt < 4; ++mt)
#pragma unroll
    for (int nt = 0; nt < 2; ++nt)
#pragma unroll
      for (int r = 0; r < 8; ++r) acc[mt][nt][r] = 0.0f;

  for (int k0 = 0; k0 < K; k0 += BK) {
    // ---- stage A tile (128x32) into LDS as bf16 ----
    if (A_F32) {
      const float* A = (const float*)Ap;
#pragma unroll
      for (int p = 0; p < 4; ++p) {
        int r = p * 32 + (tid >> 3);
        int c = (tid & 7) * 4;
        float4 v = *(const float4*)(A + (size_t)(m0 + r) * K + k0 + c);
        bf16x4 bv;
        bv[0] = (__bf16)v.x; bv[1] = (__bf16)v.y;
        bv[2] = (__bf16)v.z; bv[3] = (__bf16)v.w;
        *(bf16x4*)&As[r][c] = bv;
      }
    } else {
      const __bf16* A = (const __bf16*)Ap;
#pragma unroll
      for (int p = 0; p < 4; ++p) {
        int r = p * 32 + (tid >> 3);
        int c = (tid & 7) * 4;
        *(bf16x4*)&As[r][c] = *(const bf16x4*)(A + (size_t)(m0 + r) * K + k0 + c);
      }
    }
    // ---- stage B tile (32x128 fp32) transposed into Bs[n][k] as bf16 ----
#pragma unroll
    for (int p = 0; p < 4; ++p) {
      int k = p * 8 + (tid >> 5);
      int n = (tid & 31) * 4;
      float4 v = *(const float4*)(Bp + (size_t)(k0 + k) * N + n0 + n);
      Bs[n + 0][k] = (__bf16)v.x;
      Bs[n + 1][k] = (__bf16)v.y;
      Bs[n + 2][k] = (__bf16)v.z;
      Bs[n + 3][k] = (__bf16)v.w;
    }
    __syncthreads();

    // CDNA5 prefetch of next B tile (-> global_prefetch_b8)
    if (k0 + BK < K) {
      const char* pf =
          (const char*)(Bp + (size_t)(k0 + BK + (tid >> 5)) * N + n0 + (tid & 31) * 4);
      __builtin_prefetch(pf, 0, 1);
    }

    v16bf bfrag[2];
#pragma unroll
    for (int nt = 0; nt < 2; ++nt) bfrag[nt] = load_b16(&Bs[wn + nt * 16][0], LDT, lane);
#pragma unroll
    for (int mt = 0; mt < 4; ++mt) {
      v16bf af = load_a16(&As[wm + mt * 16][0], LDT, lane);
#pragma unroll
      for (int nt = 0; nt < 2; ++nt) acc[mt][nt] = wmma_bf16(af, bfrag[nt], acc[mt][nt]);
    }
    __syncthreads();
  }

#pragma unroll
  for (int mt = 0; mt < 4; ++mt)
#pragma unroll
    for (int nt = 0; nt < 2; ++nt)
#pragma unroll
      for (int r = 0; r < 8; ++r) {
        int m = m0 + wm + mt * 16 + r + hf * 8;
        int n = n0 + wn + nt * 16 + nl;
        float v = acc[mt][nt][r];
        if (OUT_BF16)
          ((__bf16*)Cp)[(size_t)m * N + n] = (__bf16)v;
        else
          ((float*)Cp)[(size_t)m * N + n] = v;
      }
}

// ---------------------------------------------------------------------------
// Kernel 2: RoPE exactly as the reference (rotation along the SEQUENCE axis)
// ---------------------------------------------------------------------------
__global__ __launch_bounds__(256) void rope_kernel(const __bf16* __restrict__ x,
                                                   __bf16* __restrict__ y) {
  long long idx = (long long)blockIdx.x * 256 + threadIdx.x;
  const int dp = (int)(idx & (HDIM / 2 - 1)); idx >>= 6;
  const int h  = (int)(idx & (NH - 1));       idx >>= 4;
  const int s  = (int)(idx & (S_LEN - 1));    idx >>= 11;
  const int b  = (int)idx;
  const int d0 = dp * 2;

  float freq = __expf(-(float)d0 * (9.210340371976184f / (float)HDIM));
  float ang = (float)s * freq;
  float sn, cs;
  __sincosf(ang, &sn, &cs);

  int srot; float sgn;
  if (s < S_LEN / 2) { srot = 2 * s + 1;           sgn = -1.0f; }
  else               { srot = 2 * (s - S_LEN / 2); sgn =  1.0f; }

  size_t p  = ((size_t)(b * S_LEN + s))    * EMB + h * HDIM + d0;
  size_t pr = ((size_t)(b * S_LEN + srot)) * EMB + h * HDIM + d0;
  float x0 = (float)x[p],  x1 = (float)x[p + 1];
  float r0 = (float)x[pr], r1 = (float)x[pr + 1];
  y[p]     = (__bf16)(x0 * cs + sgn * r0 * sn);
  y[p + 1] = (__bf16)(x1 * cs + sgn * r1 * sn);
}

// ---------------------------------------------------------------------------
// Kernel 3: flash-style attention with TDM staging of Q and K tiles.
// Q tile: 64x128 bf16 rows -> LDS pitch 136 elems (pad 16B every 256B row:
//   pad_interval=5 (64 DW), pad_amount=3 (4 DW)). K tile: 32x128, same pads.
// ---------------------------------------------------------------------------
__global__ __launch_bounds__(256) void attn_kernel(const __bf16* __restrict__ Q,
                                                   const __bf16* __restrict__ Kb,
                                                   const __bf16* __restrict__ V,
                                                   const float* __restrict__ mask,
                                                   __bf16* __restrict__ O) {
  __shared__ __align__(16) __bf16 Qs[64][136];
  __shared__ __align__(16) __bf16 Ks[32][136];
  __shared__ __align__(16) __bf16 Vt[HDIM][40];   // V transposed: [d][key]
  __shared__ __align__(16) float  Ss[64][40];     // scores, then exp(P) f32
  __shared__ float rmax[64], rsum[64], ralpha[64];

  const int tid  = threadIdx.x;
  const int lane = tid & 31;
  const int w    = tid >> 5;
  const int bh   = blockIdx.y;
  const int b    = bh >> 4;
  const int h    = bh & (NH - 1);
  const int s0   = blockIdx.x * 64;
  const int mt   = (w & 3) * 16;
  const int dh   = (w >> 2) * 64;
  const int hf   = lane >> 4;
  const int nl   = lane & 15;

  const unsigned ldsQ = (unsigned)(unsigned long long)(void*)&Qs[0][0];
  const unsigned ldsK = (unsigned)(unsigned long long)(void*)&Ks[0][0];

  // ---- Q tile (64x128) via Tensor Data Mover, issued by wave 0 ----
  if (w == 0) {
    tdm_load_2d_bf16(ldsQ,
                     Q + ((size_t)(b * S_LEN + s0)) * EMB + h * HDIM,
                     HDIM, 64, EMB, /*pad_interval=*/5, /*pad_amount=*/3);
  }
  if (tid < 64) { rmax[tid] = -1e30f; rsum[tid] = 0.0f; }

  v8f acc[4];
#pragma unroll
  for (int nt = 0; nt < 4; ++nt)
#pragma unroll
    for (int r = 0; r < 8; ++r) acc[nt][r] = 0.0f;

  if (w == 0) __builtin_amdgcn_s_wait_tensorcnt(0);
  __syncthreads();

  for (int j = 0; j < S_LEN; j += 32) {
    // ---- K tile (32x128) via TDM (wave 0) ----
    if (w == 0) {
      tdm_load_2d_bf16(ldsK,
                       Kb + ((size_t)(b * S_LEN + j)) * EMB + h * HDIM,
                       HDIM, 32, EMB, /*pad_interval=*/5, /*pad_amount=*/3);
    }
    // ---- V tile transposed Vt[d][key] (manual; TDM cannot transpose) ----
    {
      int krow = tid >> 3;
      int db   = (tid & 7) * 16;
      const __bf16* src = V + ((size_t)(b * S_LEN + j + krow)) * EMB + h * HDIM + db;
#pragma unroll
      for (int i = 0; i < 16; ++i) Vt[db + i][krow] = src[i];
    }
    if (w == 0) __builtin_amdgcn_s_wait_tensorcnt(0);
    __syncthreads();

    // ---- scores: (Q K^T + mask) * 64 ----
    {
      const int sm = (w & 3) * 16;
      const int sn = (w >> 2) * 16;
      v8f sacc;
#pragma unroll
      for (int r = 0; r < 8; ++r) sacc[r] = 0.0f;
#pragma unroll
      for (int ks = 0; ks < 4; ++ks) {
        v16bf af = load_a16(&Qs[sm][ks * 32], 136, lane);
        v16bf bf = load_b16(&Ks[sn][ks * 32], 136, lane);
        sacc = wmma_bf16(af, bf, sacc);
      }
#pragma unroll
      for (int r = 0; r < 8; ++r) {
        int mrow = sm + r + hf * 8;
        int kcol = sn + nl;
        float mval = mask[(size_t)(s0 + mrow) * S_LEN + j + kcol];
        Ss[mrow][kcol] = (sacc[r] + mval) * 64.0f;  // norm_scale = sqrt(2E)
      }
    }
    __syncthreads();

    // ---- online softmax bookkeeping (one thread per query row) ----
    if (tid < 64) {
      float mold = rmax[tid], mx = mold;
#pragma unroll 8
      for (int c = 0; c < 32; ++c) mx = fmaxf(mx, Ss[tid][c]);
      float a = __expf(mold - mx);
      float sum = 0.0f;
#pragma unroll 8
      for (int c = 0; c < 32; ++c) {
        float e = __expf(Ss[tid][c] - mx);
        Ss[tid][c] = e;
        sum += e;
      }
      rmax[tid]   = mx;
      rsum[tid]   = rsum[tid] * a + sum;
      ralpha[tid] = a;
    }
    __syncthreads();

    // ---- rescale O, then O += P * V ----
    {
#pragma unroll
      for (int nt = 0; nt < 4; ++nt)
#pragma unroll
        for (int r = 0; r < 8; ++r) acc[nt][r] *= ralpha[mt + r + hf * 8];

      v16bf pf = load_a16_f32(&Ss[mt][0], 40, lane);
#pragma unroll
      for (int nt = 0; nt < 4; ++nt) {
        v16bf vf = load_b16(&Vt[dh + nt * 16][0], 40, lane);
        acc[nt] = wmma_bf16(pf, vf, acc[nt]);
      }
    }
    __syncthreads();
  }

  // ---- normalize and store (b, s, h, d) as bf16 ----
#pragma unroll
  for (int nt = 0; nt < 4; ++nt)
#pragma unroll
    for (int r = 0; r < 8; ++r) {
      int mrow = mt + r + hf * 8;
      float v = acc[nt][r] / rsum[mrow];
      O[((size_t)(b * S_LEN + s0 + mrow)) * EMB + h * HDIM + dh + nt * 16 + nl] = (__bf16)v;
    }
}

// ---------------------------------------------------------------------------
// Host launcher
// ---------------------------------------------------------------------------
extern "C" void kernel_launch(void* const* d_in, const int* in_sizes, int n_in,
                              void* d_out, int out_size, void* d_ws, size_t ws_size,
                              hipStream_t stream) {
  const float* x    = (const float*)d_in[0];
  const float* mask = (const float*)d_in[1];
  const float* wq   = (const float*)d_in[2];
  const float* wk   = (const float*)d_in[3];
  const float* wv   = (const float*)d_in[4];
  const float* wo   = (const float*)d_in[5];
  float* out = (float*)d_out;

  const size_t TSZ = (size_t)BATCH * S_LEN * EMB * sizeof(__bf16);  // 16 MB
  char* ws = (char*)d_ws;
  __bf16* qraw = (__bf16*)(ws + 0 * TSZ);
  __bf16* kraw = (__bf16*)(ws + 1 * TSZ);
  __bf16* vbuf = (__bf16*)(ws + 2 * TSZ);
  __bf16* qr   = (__bf16*)(ws + 3 * TSZ);
  __bf16* kr   = (__bf16*)(ws + 4 * TSZ);
  __bf16* sa   = qraw;  // qraw retired after RoPE; reuse for attention output

  const int M = BATCH * S_LEN;  // 4096
  dim3 gg(EMB / BN, M / BM);    // (16, 32)

  gemm_tile<true, true><<<gg, 256, 0, stream>>>(x, wq, qraw, M, EMB, EMB);
  gemm_tile<true, true><<<gg, 256, 0, stream>>>(x, wk, kraw, M, EMB, EMB);
  gemm_tile<true, true><<<gg, 256, 0, stream>>>(x, wv, vbuf, M, EMB, EMB);

  const int rope_blocks = (BATCH * S_LEN * NH * (HDIM / 2)) / 256;  // 16384
  rope_kernel<<<rope_blocks, 256, 0, stream>>>(qraw, qr);
  rope_kernel<<<rope_blocks, 256, 0, stream>>>(kraw, kr);

  attn_kernel<<<dim3(S_LEN / 64, BATCH * NH), 256, 0, stream>>>(qr, kr, vbuf, mask, sa);

  gemm_tile<false, false><<<gg, 256, 0, stream>>>(sa, wo, out, M, EMB, EMB);

  (void)in_sizes; (void)n_in; (void)out_size; (void)ws_size;
}
